// QLSTMHybrid_65481071407722
// MI455X (gfx1250) — matrix operand order repaired and checked by reading, not compiled, not run on hardware
//
#include <hip/hip_runtime.h>
#include <hip/hip_bf16.h>
#include <math.h>

// ---------------------------------------------------------------------------
// QLSTM hybrid: conv1d encoder (bf16 WMMA GEMM) + persistent-kernel LayerNorm
// LSTM scan (bf16 WMMA, fp32 accumulate, one grid barrier per step).
// Target: gfx1250 (MI455X), wave32, V_WMMA_F32_16X16X32_BF16.
// ---------------------------------------------------------------------------

typedef __attribute__((ext_vector_type(16))) __bf16 v16bf;
typedef __attribute__((ext_vector_type(8)))  __bf16 bf16x8;
typedef __attribute__((ext_vector_type(4)))  __bf16 bf16x4;
typedef __attribute__((ext_vector_type(8)))  float  v8f;
typedef __attribute__((ext_vector_type(4)))  float  f32x4;

union V16 { v16bf v; f32x4 f[2]; };

#define S_LEN 1024
#define BATCH 64
#define IDIM  256
#define HDIM  512
#define K2H   1024   /* 2H  */
#define G4H   2048   /* 4H  */
#define CONVK 768    /* 3*I */
#define LN_EPS 1e-5f

#define NWG_LSTM 16

// LDS row strides (halves), padded so 16 consecutive rows hit distinct banks
#define LDA_CONV 776   /* 768 + 8  -> 1552B rows */
#define LDA_LSTM 1032  /* 1024 + 8 -> 2064B rows */

// workspace layout (bytes)
#define OFF_ENCB 0ULL                                              /* bf16 [S][B][H]   64 MB  */
#define OFF_WB   (OFF_ENCB + (size_t)S_LEN * BATCH * HDIM * 2)     /* bf16 [4H][2H]     4 MB  */
#define OFF_CW   (OFF_WB   + (size_t)G4H * K2H * 2)                /* bf16 [H][768]   768 KB  */
#define OFF_SX   (OFF_CW   + (size_t)HDIM * CONVK * 2)             /* f32  [S*B]      256 KB  */
#define OFF_SXX  (OFF_SX   + (size_t)S_LEN * BATCH * 4)            /* f32  [S*B]      256 KB  */
#define OFF_HBUF (OFF_SXX  + (size_t)S_LEN * BATCH * 4)            /* f32  [2][B][H]  256 KB  */
#define OFF_CBUF (OFF_HBUF + (size_t)2 * BATCH * HDIM * 4)         /* f32  [B][H]     128 KB  */
#define OFF_BAR  (OFF_CBUF + (size_t)BATCH * HDIM * 4)             /* int  [S]          4 KB  */
// total ~73 MB

// ---------------------------------------------------------------------------
__global__ __launch_bounds__(256) void k_init(float* hbuf, float* cbuf, int* bar) {
  int i = blockIdx.x * 256 + threadIdx.x;
  if (i < 2 * BATCH * HDIM) hbuf[i] = 0.f;
  if (i < BATCH * HDIM)     cbuf[i] = 0.f;
  if (i < S_LEN)            bar[i]  = 0;
}

// Pack W = concat(Wf,Wi,Wg,Wo) rows, convert f32 -> bf16, row-major [4H][2H]
__global__ __launch_bounds__(256) void k_cvt_w(const float* __restrict__ Wf,
                                               const float* __restrict__ Wi,
                                               const float* __restrict__ Wg,
                                               const float* __restrict__ Wo,
                                               __bf16* __restrict__ Wb) {
  int idx = blockIdx.x * 256 + threadIdx.x;    // [0, 4H*2H)
  int row = idx >> 10;
  int k   = idx & 1023;
  const float* src = (row < 512) ? Wf : (row < 1024) ? Wi : (row < 1536) ? Wg : Wo;
  int r = row & 511;
  Wb[idx] = (__bf16)src[r * K2H + k];
}

// conv_w (H,I,3) -> Cw[h][k*I + i] bf16 (im2col-friendly B matrix)
__global__ __launch_bounds__(256) void k_cvt_cw(const float* __restrict__ cw,
                                                __bf16* __restrict__ Cw) {
  int idx = blockIdx.x * 256 + threadIdx.x;    // [0, H*768)
  int h = idx / CONVK;
  int c = idx % CONVK;
  int k = c >> 8;
  int i = c & 255;
  Cw[idx] = (__bf16)cw[h * CONVK + i * 3 + k];
}

// ---------------------------------------------------------------------------
// Conv encoder: for each s, enc[s] (64x512) = A(64x768) @ Cw^T via WMMA bf16.
// A block = [inputs[s-1] | inputs[s] | inputs[s+1]] (zero padded at ends).
// grid.x = S, 256 threads = 8 waves; wave w -> 64 output cols, 4 M-tiles.
// ---------------------------------------------------------------------------
__global__ __launch_bounds__(256) void k_conv(const float* __restrict__ in,
                                              const __bf16* __restrict__ Cw,
                                              __bf16* __restrict__ encb) {
  __shared__ __bf16 Ash[64 * LDA_CONV];
  const int s = blockIdx.x;
  const int t = threadIdx.x;

  // stage A (bf16) into LDS, 4 elements per iteration (float4 in, bf16x4 out)
  for (int e = 0; e < (64 * CONVK) / (256 * 4); ++e) {
    int idx4 = e * 256 + t;
    int b = idx4 / 192, cc = (idx4 % 192) * 4;
    int slab = cc >> 8, i = cc & 255;
    int sp = s + slab - 1;
    f32x4 v = {0.f, 0.f, 0.f, 0.f};
    if (sp >= 0 && sp < S_LEN)
      v = *(const f32x4*)(in + ((size_t)sp * BATCH + b) * IDIM + i);
    bf16x4 o;
    for (int q = 0; q < 4; ++q) o[q] = (__bf16)v[q];
    *(bf16x4*)(Ash + b * LDA_CONV + cc) = o;
  }
  __syncthreads();

  const int wave = t >> 5, lane = t & 31;
  const int half = lane >> 4, lc = lane & 15;
  const int nbase = wave * 64;

  v8f acc[4][4];
  for (int m = 0; m < 4; ++m)
    for (int j = 0; j < 4; ++j)
      for (int r = 0; r < 8; ++r) acc[m][j][r] = 0.f;

  for (int ks = 0; ks < CONVK / 32; ++ks) {
    V16 bm[4];
    for (int j = 0; j < 4; ++j) {           // B: lane holds W row n, K = 16*half + 0..15
      int n = nbase + j * 16 + lc;
      const f32x4* p = (const f32x4*)(Cw + (size_t)n * CONVK + ks * 32 + half * 16);
      bm[j].f[0] = p[0];
      bm[j].f[1] = p[1];
    }
    for (int m = 0; m < 4; ++m) {           // A: lane holds row M, K = 16*half + 0..15
      int arow = m * 16 + lc;
      V16 am;
      am.f[0] = *(const f32x4*)(Ash + arow * LDA_CONV + ks * 32 + half * 16);
      am.f[1] = *(const f32x4*)(Ash + arow * LDA_CONV + ks * 32 + half * 16 + 8);
      for (int j = 0; j < 4; ++j)
        acc[m][j] = __builtin_amdgcn_wmma_f32_16x16x32_bf16(
            false, am.v, false, bm[j].v, (short)0, acc[m][j], false, false);
    }
  }

  // store enc as bf16: C layout -> (M = r + 8*half, N = lc)
  for (int m = 0; m < 4; ++m)
    for (int j = 0; j < 4; ++j) {
      int n = nbase + j * 16 + lc;
      for (int r = 0; r < 8; ++r) {
        int bi = m * 16 + half * 8 + r;
        encb[((size_t)s * BATCH + bi) * HDIM + n] = (__bf16)acc[m][j][r];
      }
    }
}

// per-(s,b) sum / sumsq of enc (the x-half of the LayerNorm statistics)
__global__ __launch_bounds__(256) void k_rowsum(const __bf16* __restrict__ encb,
                                                float* __restrict__ sx,
                                                float* __restrict__ sxx) {
  int row  = blockIdx.x * 8 + (threadIdx.x >> 5);
  int lane = threadIdx.x & 31;
  const __bf16* p = encb + (size_t)row * HDIM;
  float s1 = 0.f, s2 = 0.f;
  for (int j = 0; j < 2; ++j) {
    bf16x8 v = *(const bf16x8*)(p + lane * 8 + j * 256);
    for (int q = 0; q < 8; ++q) {
      float x = (float)v[q];
      s1 += x; s2 += x * x;
    }
  }
  for (int off = 16; off > 0; off >>= 1) {
    s1 += __shfl_xor(s1, off, 32);
    s2 += __shfl_xor(s2, off, 32);
  }
  if (lane == 0) { sx[row] = s1; sxx[row] = s2; }
}

// ---------------------------------------------------------------------------
// Persistent LSTM scan. 16 WGs x 256 threads. Each WG owns a 32-wide hidden
// slice; wave (mg,hs) computes 1 M-tile x 4 gate-tiles for a 16-wide hidden
// sub-slice, so f/i/g/o for each (b,h) are co-resident in one lane's VGPRs.
// One grid barrier per step (h double-buffered). GEMM fragments are
// double-buffered so next-k loads overlap current-k WMMAs.
// ---------------------------------------------------------------------------
__global__ __launch_bounds__(256) void k_lstm(const __bf16* __restrict__ encb,
                                              const __bf16* __restrict__ Wb,
                                              const float* __restrict__ sx,
                                              const float* __restrict__ sxx,
                                              const float* __restrict__ lnw,
                                              const float* __restrict__ lnb,
                                              const float* __restrict__ bfp,
                                              const float* __restrict__ bip,
                                              const float* __restrict__ bgp,
                                              const float* __restrict__ bop,
                                              float* __restrict__ hbuf,
                                              float* __restrict__ cbuf,
                                              int* __restrict__ bar,
                                              float* __restrict__ out) {
  __shared__ __bf16 Aln[64 * LDA_LSTM];            // normalized [x_t | h_t], bf16
  __shared__ float smu[64], sinv[64];
  __shared__ float slnw[K2H], slnb[K2H];

  const int t = threadIdx.x;
  const int wave = t >> 5, lane = t & 31;
  const int half = lane >> 4, lc = lane & 15;
  const int mg = wave & 3;                          // M tile index (16 batch rows)
  const int hs = wave >> 2;                         // hidden 16-slice within WG
  const int nh = blockIdx.x * 32 + hs * 16 + lc;    // hidden index [0,512)

  for (int e = 0; e < 4; ++e) {                     // cache LN params in LDS
    int k = e * 256 + t;
    slnw[k] = lnw[k];
    slnb[k] = lnb[k];
  }
  const float bf_ = bfp[nh], bi_ = bip[nh], bg_ = bgp[nh], bo_ = bop[nh];
  const __bf16* Wrow[4];
  for (int g = 0; g < 4; ++g) Wrow[g] = Wb + ((size_t)(g * HDIM + nh)) * K2H;
  const __bf16* Arow = Aln + (mg * 16 + lc) * LDA_LSTM;  // this lane's A row
  __syncthreads();

  for (int s = 0; s < S_LEN; ++s) {
    const int cur = s & 1, nxt = cur ^ 1;
    const float* h = hbuf + cur * BATCH * HDIM;

    // --- LayerNorm stats (enc-half precomputed; h-half reduced here) ---
    for (int rr = 0; rr < 8; ++rr) {
      int b = wave * 8 + rr;
      const float* hp = h + b * HDIM;
      float s1 = 0.f, s2 = 0.f;
      for (int j = 0; j < 4; ++j) {
        f32x4 v = *(const f32x4*)(hp + lane * 4 + j * 128);
        for (int q = 0; q < 4; ++q) { s1 += v[q]; s2 += v[q] * v[q]; }
      }
      for (int off = 16; off > 0; off >>= 1) {
        s1 += __shfl_xor(s1, off, 32);
        s2 += __shfl_xor(s2, off, 32);
      }
      if (lane == 0) {
        float tot  = sx[s * BATCH + b] + s1;
        float tot2 = sxx[s * BATCH + b] + s2;
        float mu   = tot * (1.f / K2H);
        float var  = tot2 * (1.f / K2H) - mu * mu;
        smu[b]  = mu;
        sinv[b] = rsqrtf(var + LN_EPS);
      }
    }
    __syncthreads();

    // --- stage normalized A (64 x 1024) as bf16 into LDS, 8-wide ---
    // chunk mapping keeps the enc-vs-h branch uniform per wave:
    // chunk-in-row = t & 127, so waves 0,1,4,5 take enc, waves 2,3,6,7 take h.
    const __bf16* encs = encb + (size_t)s * BATCH * HDIM;
    for (int e = 0; e < (64 * K2H) / (256 * 8); ++e) {
      int idx8 = e * 256 + t;
      int b = idx8 >> 7, kc = (idx8 & 127) * 8;
      float mu = smu[b], inv = sinv[b];
      f32x4 w0 = *(const f32x4*)(slnw + kc), w1 = *(const f32x4*)(slnw + kc + 4);
      f32x4 c0 = *(const f32x4*)(slnb + kc), c1 = *(const f32x4*)(slnb + kc + 4);
      float vals[8];
      if (kc < HDIM) {
        bf16x8 ev = *(const bf16x8*)(encs + b * HDIM + kc);
        for (int q = 0; q < 8; ++q) vals[q] = (float)ev[q];
      } else {
        const f32x4* hv = (const f32x4*)(h + b * HDIM + (kc - HDIM));
        f32x4 h0 = hv[0], h1 = hv[1];
        for (int q = 0; q < 4; ++q) { vals[q] = h0[q]; vals[q + 4] = h1[q]; }
      }
      bf16x8 o;
      for (int q = 0; q < 4; ++q) {
        o[q]     = (__bf16)((vals[q]     - mu) * inv * w0[q] + c0[q]);
        o[q + 4] = (__bf16)((vals[q + 4] - mu) * inv * w1[q] + c1[q]);
      }
      *(bf16x8*)(Aln + b * LDA_LSTM + kc) = o;
    }
    __syncthreads();

    // --- GEMM: 1 M-tile x 4 gates, K = 1024 (32 WMMA k-steps), ---
    // --- double-buffered A (LDS) and B (global) fragments        ---
    v8f acc[4];
    for (int g = 0; g < 4; ++g)
      for (int r = 0; r < 8; ++r) acc[g][r] = 0.f;

    V16 am[2], bm[2][4];
    {
      const int c = half * 16;
      for (int g = 0; g < 4; ++g) {
        const f32x4* p = (const f32x4*)(Wrow[g] + c);
        bm[0][g].f[0] = p[0]; bm[0][g].f[1] = p[1];
      }
      am[0].f[0] = *(const f32x4*)(Arow + half * 16);
      am[0].f[1] = *(const f32x4*)(Arow + half * 16 + 8);
    }
    for (int ks = 0; ks < 32; ++ks) {
      const int cu = ks & 1, nx = cu ^ 1;
      if (ks < 31) {
        const int c = (ks + 1) * 32;
        for (int g = 0; g < 4; ++g) {
          const f32x4* p = (const f32x4*)(Wrow[g] + c + half * 16);
          bm[nx][g].f[0] = p[0]; bm[nx][g].f[1] = p[1];
          if (ks < 24)  // global_prefetch_b8 of weights 8 k-steps ahead
            __builtin_prefetch(Wrow[g] + (ks + 8) * 32 + half * 16, 0, 1);
        }
        am[nx].f[0] = *(const f32x4*)(Arow + c + half * 16);
        am[nx].f[1] = *(const f32x4*)(Arow + c + half * 16 + 8);
      }
      for (int g = 0; g < 4; ++g)
        acc[g] = __builtin_amdgcn_wmma_f32_16x16x32_bf16(
            false, am[cu].v, false, bm[cu][g].v, (short)0, acc[g], false, false);
    }

    // --- cell update entirely in registers (f/i/g/o co-located) ---
    float* hn = hbuf + nxt * BATCH * HDIM;
    for (int r = 0; r < 8; ++r) {
      int b = mg * 16 + half * 8 + r;
      float fv = acc[0][r] + bf_;
      float iv = acc[1][r] + bi_;
      float gv = acc[2][r] + bg_;
      float ov = acc[3][r] + bo_;
      float sf = 1.f / (1.f + __expf(-fv));
      float si = 1.f / (1.f + __expf(-iv));
      float so = 1.f / (1.f + __expf(-ov));
      float tg = tanhf(gv);
      int ci = b * HDIM + nh;
      float cn = sf * cbuf[ci] + si * tg;
      float hv = so * tanhf(cn);
      cbuf[ci] = cn;
      hn[ci] = hv;
      out[(size_t)s * (BATCH * HDIM) + ci] = hv;
      if (s == S_LEN - 1) {
        out[(size_t)S_LEN * BATCH * HDIM + ci] = hv;                    // hx
        out[(size_t)S_LEN * BATCH * HDIM + BATCH * HDIM + ci] = cn;     // cx
      }
    }

    // --- grid barrier (release h writes, acquire before next-step reads) ---
    __syncthreads();
    if (t == 0) {
      __hip_atomic_fetch_add(&bar[s], 1, __ATOMIC_RELEASE, __HIP_MEMORY_SCOPE_AGENT);
      while (__hip_atomic_load(&bar[s], __ATOMIC_ACQUIRE, __HIP_MEMORY_SCOPE_AGENT) <
             (int)gridDim.x)
        __builtin_amdgcn_s_sleep(1);
    }
    __syncthreads();
    __threadfence();
  }
}

// ---------------------------------------------------------------------------
extern "C" void kernel_launch(void* const* d_in, const int* in_sizes, int n_in,
                              void* d_out, int out_size, void* d_ws, size_t ws_size,
                              hipStream_t stream) {
  const float* inputs = (const float*)d_in[0];
  const float* conv_w = (const float*)d_in[1];
  const float* ln_w   = (const float*)d_in[2];
  const float* ln_b   = (const float*)d_in[3];
  const float* Wf     = (const float*)d_in[4];
  const float* bf_    = (const float*)d_in[5];
  const float* Wi     = (const float*)d_in[6];
  const float* bi_    = (const float*)d_in[7];
  const float* Wg     = (const float*)d_in[8];
  const float* bg_    = (const float*)d_in[9];
  const float* Wo     = (const float*)d_in[10];
  const float* bo_    = (const float*)d_in[11];
  float* out = (float*)d_out;

  char* ws = (char*)d_ws;
  __bf16* encb = (__bf16*)(ws + OFF_ENCB);
  __bf16* Wb   = (__bf16*)(ws + OFF_WB);
  __bf16* Cw   = (__bf16*)(ws + OFF_CW);
  float*  sx   = (float*)(ws + OFF_SX);
  float*  sxx  = (float*)(ws + OFF_SXX);
  float*  hbuf = (float*)(ws + OFF_HBUF);
  float*  cbuf = (float*)(ws + OFF_CBUF);
  int*    bar  = (int*)(ws + OFF_BAR);

  k_init<<<256, 256, 0, stream>>>(hbuf, cbuf, bar);
  k_cvt_w<<<(G4H * K2H) / 256, 256, 0, stream>>>(Wf, Wi, Wg, Wo, Wb);
  k_cvt_cw<<<(HDIM * CONVK) / 256, 256, 0, stream>>>(conv_w, Cw);
  k_conv<<<S_LEN, 256, 0, stream>>>(inputs, Cw, encb);
  k_rowsum<<<(S_LEN * BATCH) / 8, 256, 0, stream>>>(encb, sx, sxx);
  k_lstm<<<NWG_LSTM, 256, 0, stream>>>(encb, Wb, sx, sxx, ln_w, ln_b,
                                       bf_, bi_, bg_, bo_, hbuf, cbuf, bar, out);
}
